// GroupedQueryAttention_55551107006926
// MI455X (gfx1250) — matrix operand order repaired and checked by reading, not compiled, not run on hardware
//
#include <hip/hip_runtime.h>
#include <hip/hip_bf16.h>

#define DI __device__ __forceinline__

constexpr int B_ = 2, T_ = 2048, HID_ = 2048;
constexpr int H_ = 16, HKV_ = 4, HD_ = 128;
constexpr int GROUPS_ = H_ / HKV_;
constexpr float SCALE_ = 0.08838834764831845f; // 128^-0.5

// padded LDS row strides (bf16 elements); all row pitches are 16B-aligned and
// map lanes to distinct bank quads (stride%64banks pattern 20 dwords)
constexpr int BP_ = 40;    // 32-K tiles padded 32->40
constexpr int KP_ = 136;   // 128-d K tile rows padded 128->136

typedef __attribute__((ext_vector_type(16))) __bf16        v16bf;
typedef __attribute__((ext_vector_type(8)))  float         v8f;
typedef __attribute__((ext_vector_type(8)))  unsigned int  v8u;

// ---------- helpers ----------

DI unsigned short f2bf(float f) {
  unsigned int u = __builtin_bit_cast(unsigned int, f);
  u += 0x7fffu + ((u >> 16) & 1u);           // round-to-nearest-even
  return (unsigned short)(u >> 16);
}

// A-matrix 16x32 bf16 fragment (ISA 7.12.2): lane m = lane%16 selects row,
// VGPR r holds K pair: koff = (r<4?0:16) + (lane/16)*8 + (r%4)*2
DI v16bf load_a_frag(const unsigned short* base, int stride) {
  const int lane = threadIdx.x & 31;
  const int m = lane & 15, g = lane >> 4;
  const unsigned short* p = base + (size_t)m * stride;
  v8u u;
#pragma unroll
  for (int r = 0; r < 8; ++r) {
    const int koff = ((r & 4) ? 16 : 0) + g * 8 + (r & 3) * 2;
    u[r] = *(const unsigned int*)(p + koff);
  }
  return __builtin_bit_cast(v16bf, u);
}

// B-matrix 32x16 bf16 fragment: lane n = lane%16 selects column,
// VGPR i holds K pair k = (i<4?0:16) + (lane/16)*8 + (i%4)*2. Source layout: src[n*stride + k].
DI v16bf load_b_frag(const unsigned short* base, int stride) {
  const int lane = threadIdx.x & 31;
  const int n = lane & 15, g = lane >> 4;
  const unsigned short* p = base + (size_t)n * stride + g * 8;
  v8u u;
#pragma unroll
  for (int i = 0; i < 8; ++i) u[i] = *(const unsigned int*)(p + ((i & 4) ? 16 : 0) + (i & 3) * 2);
  return __builtin_bit_cast(v16bf, u);
}

DI v8f wmma_bf16(v16bf a, v16bf b, v8f c) {
  return __builtin_amdgcn_wmma_f32_16x16x32_bf16(false, a, false, b, (short)0, c,
                                                 false, false);
}

// CDNA5 async copy: 16B global -> LDS, tracked by ASYNCcnt
DI void async_b128(unsigned lds_off, const void* g) {
  asm volatile("global_load_async_to_lds_b128 %0, %1, off"
               :: "v"(lds_off), "v"((unsigned long long)g)
               : "memory");
}
#define WAIT_ASYNC0() asm volatile("s_wait_asynccnt 0x0" ::: "memory")
#define WAIT_DS0()    asm volatile("s_wait_dscnt 0x0" ::: "memory")

DI unsigned lds_off_of(const void* p) { return (unsigned)(uintptr_t)p; }

// ---------- elementwise prep kernels ----------

__global__ void cvt_bf16_kernel(const float* __restrict__ s,
                                unsigned short* __restrict__ d, int n) {
  int i = blockIdx.x * blockDim.x + threadIdx.x;
  if (i < n) d[i] = f2bf(s[i]);
}

// w [K,N] f32 -> wt [N,K] bf16
__global__ void wt_transpose_kernel(const float* __restrict__ w,
                                    unsigned short* __restrict__ wt, int K, int N) {
  int i = blockIdx.x * blockDim.x + threadIdx.x;
  if (i >= K * N) return;
  int n = i % N, k = i / N;
  wt[(size_t)n * K + k] = f2bf(w[i]);
}

// src [B,T,heads,HD] f32 -> dst [B,heads,T,HD] bf16 with rotate-half RoPE
__global__ void rope_kernel(const float* __restrict__ src,
                            unsigned short* __restrict__ dst, int heads) {
  int i = blockIdx.x * blockDim.x + threadIdx.x;
  int total = B_ * T_ * heads * (HD_ / 2);
  if (i >= total) return;
  int d = i % 64;
  int h = (i / 64) % heads;
  int t = (i / (64 * heads)) % T_;
  int b = i / (64 * heads * T_);
  const float* p = src + ((size_t)(b * T_ + t) * heads + h) * HD_;
  float x1 = p[d], x2 = p[d + 64];
  float inv = __expf(-9.210340371976184f * (float)d * (1.0f / 64.0f)); // 10000^(-d/64)
  float ang = (float)t * inv;
  float sn, cs;
  __sincosf(ang, &sn, &cs);
  unsigned short* q = dst + ((size_t)(b * heads + h) * T_ + t) * HD_;
  q[d]      = f2bf(x1 * cs - x2 * sn);
  q[d + 64] = f2bf(x2 * cs + x1 * sn);
}

// src [B,T,HKV,HD] f32 -> dst [B,HKV,HD,T] bf16
__global__ void v_transpose_kernel(const float* __restrict__ src,
                                   unsigned short* __restrict__ dst) {
  int i = blockIdx.x * blockDim.x + threadIdx.x;
  int total = B_ * T_ * HKV_ * HD_;
  if (i >= total) return;
  int d = i % HD_;
  int h = (i / HD_) % HKV_;
  int t = (i / (HD_ * HKV_)) % T_;
  int b = i / (HD_ * HKV_ * T_);
  dst[((size_t)(b * HKV_ + h) * HD_ + d) * T_ + t] = f2bf(src[i]);
}

// ---------- WMMA GEMM: C[M,N](f32) = A[M,K](bf16,row) * Bt[N,K](bf16)^T ------
// 8 waves/workgroup, each wave computes 32(M) x 64(N) -> 256x64 per block.
// B tile (64N x 32K) async-staged in LDS (double buffered, padded rows);
// every B fragment feeds two WMMAs (both M sub-tiles). A fragments are
// register-pipelined one stage ahead.

__global__ void gemm_bf16_kernel(const unsigned short* __restrict__ A,
                                 const unsigned short* __restrict__ Bt,
                                 float* __restrict__ C, int M, int N, int K) {
  __shared__ unsigned short Btile[2][64 * BP_];   // ~10 KB
  const int wid = threadIdx.x >> 5;
  const int n0 = blockIdx.x * 64;
  const int m0 = blockIdx.y * 256 + wid * 32;
  const unsigned short* A0 = A + (size_t)m0 * K;
  const unsigned short* A1 = A0 + (size_t)16 * K;

  // one 16B chunk per thread: 64 rows x 32K = 4KB per stage
  auto stage_b = [&](int buf, int kc) {
    const int t = threadIdx.x;
    const int row = t >> 2, cc = t & 3;
    async_b128(lds_off_of(&Btile[buf][row * BP_ + cc * 8]),
               Bt + (size_t)(n0 + row) * K + kc + cc * 8);
  };

  v8f acc0[4] = {}, acc1[4] = {};
  stage_b(0, 0);
  v16bf a0 = load_a_frag(A0, K);
  v16bf a1 = load_a_frag(A1, K);
  WAIT_ASYNC0();
  __syncthreads();

  int buf = 0;
  for (int kc = 0; kc < K; kc += 32) {
    const bool has_next = (kc + 32) < K;
    if (has_next) stage_b(buf ^ 1, kc + 32);
    v16bf a0n = a0, a1n = a1;
    if (has_next) {
      a0n = load_a_frag(A0 + kc + 32, K);
      a1n = load_a_frag(A1 + kc + 32, K);
    }
#pragma unroll
    for (int j = 0; j < 4; ++j) {
      v16bf bj = load_b_frag(&Btile[buf][(j * 16) * BP_], BP_);
      acc0[j] = wmma_bf16(a0, bj, acc0[j]);
      acc1[j] = wmma_bf16(a1, bj, acc1[j]);
    }
    a0 = a0n; a1 = a1n;
    WAIT_ASYNC0();
    __syncthreads();
    buf ^= 1;
  }

  const int lane = threadIdx.x & 31;
  const int g = lane >> 4, nc = lane & 15;
#pragma unroll
  for (int j = 0; j < 4; ++j)
#pragma unroll
    for (int r = 0; r < 8; ++r) {
      C[(size_t)(m0 + r + g * 8) * N + n0 + j * 16 + nc]      = acc0[j][r];
      C[(size_t)(m0 + 16 + r + g * 8) * N + n0 + j * 16 + nc] = acc1[j][r];
    }
}

// ---------- Flash attention ----------
// 4 waves per block share (b,h) and 4 consecutive 16-query tiles; K/V 32-key
// chunks async-staged in LDS (double buffered, padded rows) and shared by all
// 4 waves. Qb [B,H,T,HD], Kb [B,HKV,T,HD], Vt [B,HKV,HD,T], attn_b [B*T,H*HD].

__global__ void flash_kernel(const unsigned short* __restrict__ Qb,
                             const unsigned short* __restrict__ Kb,
                             const unsigned short* __restrict__ Vt,
                             unsigned short* __restrict__ attn_b) {
  __shared__ unsigned short Kl[2][32 * KP_];    // [key][d], padded
  __shared__ unsigned short Vl[2][HD_ * BP_];   // [d][key], padded
  __shared__ unsigned short Pl[4][16 * BP_];    // per-wave P tile, padded

  const int qtg = blockIdx.x, h = blockIdx.y, b = blockIdx.z;
  const int wid = threadIdx.x >> 5;
  const int qt = qtg * 4 + wid;
  const int kvh = h / GROUPS_;
  const int lane = threadIdx.x & 31;
  const int g = lane >> 4, nc = lane & 15;

  const unsigned short* Qbase = Qb + ((size_t)(b * H_ + h) * T_ + qt * 16) * HD_;
  const unsigned short* Kbase = Kb + (size_t)(b * HKV_ + kvh) * T_ * HD_;
  const unsigned short* Vbase = Vt + (size_t)(b * HKV_ + kvh) * HD_ * T_;

  auto stage_kv = [&](int buf, int k0) {
    const int t = threadIdx.x;            // 0..127
#pragma unroll
    for (int j = 0; j < 4; ++j) {         // K chunk: 32 rows x 128 d
      int i = t + j * 128;                // 0..511
      int row = i >> 4, c = i & 15;
      async_b128(lds_off_of(&Kl[buf][row * KP_ + c * 8]),
                 Kbase + (size_t)(k0 + row) * HD_ + c * 8);
    }
#pragma unroll
    for (int j = 0; j < 4; ++j) {         // V chunk: 128 d x 32 keys
      int i = t + j * 128;
      int row = i >> 2, c = i & 3;
      async_b128(lds_off_of(&Vl[buf][row * BP_ + c * 8]),
                 Vbase + (size_t)row * T_ + k0 + c * 8);
    }
  };

  // Q tile in registers: 4 x (16x32) fragments covering HD=128
  v16bf aq[4];
#pragma unroll
  for (int c = 0; c < 4; ++c) aq[c] = load_a_frag(Qbase + c * 32, HD_);

  v8f o[8] = {};
  float mrow[8], lrow[8];
#pragma unroll
  for (int r = 0; r < 8; ++r) { mrow[r] = -1e30f; lrow[r] = 0.f; }

  // group-wide causal chunk count: ((qtg*4+3)*16 + 15)/32 + 1 == 2*qtg + 2
  const int nch = 2 * qtg + 2;
  stage_kv(0, 0);
  WAIT_ASYNC0();
  __syncthreads();

  int buf = 0;
  for (int ch = 0; ch < nch; ++ch) {
    const int k0 = ch * 32;
    if (ch + 1 < nch) stage_kv(buf ^ 1, k0 + 32);

    // S = Q * K^T from LDS K tile (both key-half fragments before the WMMAs)
    v8f s0 = {}, s1 = {};
#pragma unroll
    for (int c = 0; c < 4; ++c) {
      v16bf b0 = load_b_frag(&Kl[buf][c * 32], KP_);
      v16bf b1 = load_b_frag(&Kl[buf][16 * KP_ + c * 32], KP_);
      s0 = wmma_bf16(aq[c], b0, s0);
      s1 = wmma_bf16(aq[c], b1, s1);
    }

    // online softmax; C-layout row m = r + 8*(lane/16), col n = lane%16
#pragma unroll
    for (int r = 0; r < 8; ++r) {
      const int q = qt * 16 + r + g * 8;
      float v0 = s0[r] * SCALE_;
      float v1 = s1[r] * SCALE_;
      if (k0 + nc > q)      v0 = -1e30f;
      if (k0 + 16 + nc > q) v1 = -1e30f;
      float mx = fmaxf(v0, v1);
#pragma unroll
      for (int d = 1; d < 16; d <<= 1) mx = fmaxf(mx, __shfl_xor(mx, d, 32));
      const float mnew  = fmaxf(mrow[r], mx);
      const float alpha = __expf(mrow[r] - mnew);
      const float p0 = __expf(v0 - mnew);
      const float p1 = __expf(v1 - mnew);
      float rs = p0 + p1;
#pragma unroll
      for (int d = 1; d < 16; d <<= 1) rs += __shfl_xor(rs, d, 32);
      lrow[r] = lrow[r] * alpha + rs;
      mrow[r] = mnew;
#pragma unroll
      for (int nt = 0; nt < 8; ++nt) o[nt][r] *= alpha;
      const int m = r + g * 8;
      Pl[wid][m * BP_ + nc]      = f2bf(p0);
      Pl[wid][m * BP_ + 16 + nc] = f2bf(p1);
    }

    WAIT_DS0();   // P tile RAW (per-wave, LDS in-order)

    // O += P(16x32) * V(32xHD); V fragment pipelined one N-tile ahead
    v16bf pa = load_a_frag(Pl[wid], BP_);
    v16bf bv = load_b_frag(&Vl[buf][0], BP_);
#pragma unroll
    for (int nt = 0; nt < 8; ++nt) {
      v16bf bvn = bv;
      if (nt < 7) bvn = load_b_frag(&Vl[buf][((nt + 1) * 16) * BP_], BP_);
      o[nt] = wmma_bf16(pa, bv, o[nt]);
      bv = bvn;
    }
    WAIT_DS0();   // WAR before next P store

    WAIT_ASYNC0();      // own prefetch done before publishing barrier
    __syncthreads();
    buf ^= 1;
  }

  // epilogue: O / l -> attn_b [B*T, H*HD] bf16
#pragma unroll
  for (int r = 0; r < 8; ++r) {
    const float inv = 1.0f / lrow[r];
    const int row = b * T_ + qt * 16 + r + g * 8;
#pragma unroll
    for (int nt = 0; nt < 8; ++nt)
      attn_b[(size_t)row * (H_ * HD_) + h * HD_ + nt * 16 + nc] =
          f2bf(o[nt][r] * inv);
  }
}

// ---------- host ----------

extern "C" void kernel_launch(void* const* d_in, const int* in_sizes, int n_in,
                              void* d_out, int out_size, void* d_ws, size_t ws_size,
                              hipStream_t stream) {
  (void)in_sizes; (void)n_in; (void)out_size; (void)ws_size;
  const float* x  = (const float*)d_in[0];
  const float* Wq = (const float*)d_in[1];
  const float* Wk = (const float*)d_in[2];
  const float* Wv = (const float*)d_in[3];
  const float* Wo = (const float*)d_in[4];
  float* out = (float*)d_out;

  char* ws = (char*)d_ws;
  size_t off = 0;
  auto alloc = [&](size_t bytes) -> void* {
    void* p = ws + off;
    off += (bytes + 255) & ~(size_t)255;
    return p;
  };

  const int BT = B_ * T_;                                  // 4096
  unsigned short* xb   = (unsigned short*)alloc((size_t)BT * HID_ * 2);
  unsigned short* WqT  = (unsigned short*)alloc((size_t)HID_ * H_ * HD_ * 2);
  unsigned short* WkT  = (unsigned short*)alloc((size_t)HID_ * HKV_ * HD_ * 2);
  unsigned short* WvT  = (unsigned short*)alloc((size_t)HID_ * HKV_ * HD_ * 2);
  unsigned short* WoT  = (unsigned short*)alloc((size_t)H_ * HD_ * HID_ * 2);
  float*          qf   = (float*)alloc((size_t)BT * H_ * HD_ * 4);
  float*          kf   = (float*)alloc((size_t)BT * HKV_ * HD_ * 4);
  float*          vf   = (float*)alloc((size_t)BT * HKV_ * HD_ * 4);
  unsigned short* Qb   = (unsigned short*)alloc((size_t)BT * H_ * HD_ * 2);
  unsigned short* Kb   = (unsigned short*)alloc((size_t)BT * HKV_ * HD_ * 2);
  unsigned short* Vt   = (unsigned short*)alloc((size_t)BT * HKV_ * HD_ * 2);
  unsigned short* attn = (unsigned short*)alloc((size_t)BT * H_ * HD_ * 2);

  const int TPB = 256;
  int n = BT * HID_;
  cvt_bf16_kernel<<<(n + TPB - 1) / TPB, TPB, 0, stream>>>(x, xb, n);
  wt_transpose_kernel<<<(HID_ * H_ * HD_ + TPB - 1) / TPB, TPB, 0, stream>>>(
      Wq, WqT, HID_, H_ * HD_);
  wt_transpose_kernel<<<(HID_ * HKV_ * HD_ + TPB - 1) / TPB, TPB, 0, stream>>>(
      Wk, WkT, HID_, HKV_ * HD_);
  wt_transpose_kernel<<<(HID_ * HKV_ * HD_ + TPB - 1) / TPB, TPB, 0, stream>>>(
      Wv, WvT, HID_, HKV_ * HD_);
  wt_transpose_kernel<<<(H_ * HD_ * HID_ + TPB - 1) / TPB, TPB, 0, stream>>>(
      Wo, WoT, H_ * HD_, HID_);

  gemm_bf16_kernel<<<dim3((H_ * HD_) / 64, BT / 256), 256, 0, stream>>>(
      xb, WqT, qf, BT, H_ * HD_, HID_);
  gemm_bf16_kernel<<<dim3((HKV_ * HD_) / 64, BT / 256), 256, 0, stream>>>(
      xb, WkT, kf, BT, HKV_ * HD_, HID_);
  gemm_bf16_kernel<<<dim3((HKV_ * HD_) / 64, BT / 256), 256, 0, stream>>>(
      xb, WvT, vf, BT, HKV_ * HD_, HID_);

  rope_kernel<<<(B_ * T_ * H_ * 64 + TPB - 1) / TPB, TPB, 0, stream>>>(qf, Qb, H_);
  rope_kernel<<<(B_ * T_ * HKV_ * 64 + TPB - 1) / TPB, TPB, 0, stream>>>(kf, Kb, HKV_);
  v_transpose_kernel<<<(B_ * T_ * HKV_ * HD_ + TPB - 1) / TPB, TPB, 0, stream>>>(vf, Vt);

  flash_kernel<<<dim3(T_ / 64, H_, B_), 128, 0, stream>>>(Qb, Kb, Vt, attn);

  gemm_bf16_kernel<<<dim3(HID_ / 64, BT / 256), 256, 0, stream>>>(
      attn, WoT, out, BT, HID_, H_ * HD_);
}
